// LocalSemanticEvidenceEncoderV3412_67199058313885
// MI455X (gfx1250) — matrix operand order repaired and checked by reading, not compiled, not run on hardware
//
#include <hip/hip_runtime.h>
#include <hip/hip_bf16.h>
#include <math.h>

#define DEV __device__ __forceinline__

// ---- problem constants ----
#define HIDc 256
#define SEMc 384
#define Bc   4
#define Mc   512
#define Tc   128
#define Hc   32
#define Uc   64

typedef __attribute__((ext_vector_type(16))) __bf16 bf16x16;
typedef __attribute__((ext_vector_type(8)))  float  f32x8;

DEV float bf2f(__bf16 h) {
    union { unsigned short s; __bf16 b; } u; u.b = h;
    union { unsigned u32; float f; } v; v.u32 = ((unsigned)u.s) << 16;
    return v.f;
}
DEV __bf16 f2bf(float f) {
    union { float f; unsigned u; } v; v.f = f;
    unsigned r = v.u + 0x7FFFu + ((v.u >> 16) & 1u);
    union { unsigned short s; __bf16 b; } o; o.s = (unsigned short)(r >> 16);
    return o.b;
}
DEV float clamp01(float x) { return fminf(fmaxf(x, 0.f), 1.f); }
DEV float wred_sum(float v) {
    #pragma unroll
    for (int i = 16; i > 0; i >>= 1) v += __shfl_xor(v, i, 32);
    return v;
}
DEV float wred_max(float v) {
    #pragma unroll
    for (int i = 16; i > 0; i >>= 1) v = fmaxf(v, __shfl_xor(v, i, 32));
    return v;
}
DEV float gelu(float x) { return 0.5f * x * (1.f + erff(x * 0.70710678118654752f)); }

// ---------- WMMA fragment loaders (ISA 7.12.2 layouts, wave32) ----------
// A: 16x32 bf16, row-major LDS. One contiguous 32B run per lane (k = 16*half + j) -> 2x ds_load_b128.
DEV bf16x16 load_a(const __bf16* A, int lda, int row0, int k0, int lane) {
    int m = lane & 15, half = lane >> 4;
    const __bf16* p = A + (size_t)(row0 + m) * lda + k0 + half * 16;
    bf16x16 a;
    #pragma unroll
    for (int j = 0; j < 16; ++j) {
        int v = j >> 1, pb = j & 1;
        int k = v * 2 + pb;
        a[j] = p[k];
    }
    return a;
}
// B from prepared fragment-layout weights: lane reads 32 contiguous bytes.
DEV bf16x16 load_b_prep(const __bf16* Wp, int kt, int ct, int ctiles, int lane) {
    const __bf16* p = Wp + (((size_t)kt * ctiles + ct) * 32 + lane) * 16;
    bf16x16 b;
    #pragma unroll
    for (int j = 0; j < 16; ++j) b[j] = p[j];
    return b;
}
// B = Mt^T where Mt is row-major bf16 LDS [N, K]; contiguous along K -> 2x ds_load_b128.
DEV bf16x16 load_bT_lds(const __bf16* Mt, int ldm, int k0, int col0, int lane) {
    int n = lane & 15, half = lane >> 4;
    int kb = k0 + half * 16;
    bf16x16 b;
    #pragma unroll
    for (int j = 0; j < 16; ++j) b[j] = Mt[(size_t)(col0 + n) * ldm + kb + j];
    return b;
}

// ---------- GEMM with prepared weights: C[RT*16, N] = act(A @ W' + bias') ----------
template <int RT>
DEV void gemm_prep(const __bf16* A, int lda, const __bf16* Wp, int ktiles,
                   const float* bias, int ctiles, int act,
                   __bf16* OutB, int ldob, bool obT,
                   float* OutF, int ldof, float* OutG, int ldog,
                   int wave, int nwaves, int lane) {
    const f32x8 zero = {0.f, 0.f, 0.f, 0.f, 0.f, 0.f, 0.f, 0.f};
    for (int ct = wave; ct < ctiles; ct += nwaves) {
        f32x8 acc[RT];
        #pragma unroll
        for (int r = 0; r < RT; ++r) acc[r] = zero;
        for (int kt = 0; kt < ktiles; ++kt) {
            bf16x16 bf = load_b_prep(Wp, kt, ct, ctiles, lane);
            #pragma unroll
            for (int r = 0; r < RT; ++r) {
                bf16x16 af = load_a(A, lda, r * 16, kt * 32, lane);
                acc[r] = __builtin_amdgcn_wmma_f32_16x16x32_bf16(
                    false, af, false, bf, (short)0, acc[r], false, false);
            }
        }
        int n = lane & 15, moff = (lane >> 4) * 8;
        float bv = bias ? bias[ct * 16 + n] : 0.f;
        #pragma unroll
        for (int r = 0; r < RT; ++r) {
            #pragma unroll
            for (int q = 0; q < 8; ++q) {
                float x = acc[r][q] + bv;
                if (act == 1) x = gelu(x);
                int row = r * 16 + moff + q, col = ct * 16 + n;
                if (OutB) {
                    if (obT) OutB[(size_t)col * ldob + row] = f2bf(x);
                    else     OutB[(size_t)row * ldob + col] = f2bf(x);
                }
                if (OutF) OutF[(size_t)row * ldof + col] = x;
                if (OutG) OutG[(size_t)row * ldog + col] = x;
            }
        }
    }
}

// GEMM with B^T in LDS (attn @ value via value^T)
template <int RT>
DEV void gemm_ldsBT(const __bf16* A, int lda, const __bf16* Mt, int ldm, int K,
                    int ctiles, float* OutF, int ldof, float* OutG, int ldog,
                    int wave, int nwaves, int lane) {
    const f32x8 zero = {0.f, 0.f, 0.f, 0.f, 0.f, 0.f, 0.f, 0.f};
    for (int ct = wave; ct < ctiles; ct += nwaves) {
        f32x8 acc[RT];
        #pragma unroll
        for (int r = 0; r < RT; ++r) acc[r] = zero;
        for (int k0 = 0; k0 < K; k0 += 32) {
            bf16x16 bf = load_bT_lds(Mt, ldm, k0, ct * 16, lane);
            #pragma unroll
            for (int r = 0; r < RT; ++r) {
                bf16x16 af = load_a(A, lda, r * 16, k0, lane);
                acc[r] = __builtin_amdgcn_wmma_f32_16x16x32_bf16(
                    false, af, false, bf, (short)0, acc[r], false, false);
            }
        }
        int n = lane & 15, moff = (lane >> 4) * 8;
        #pragma unroll
        for (int r = 0; r < RT; ++r) {
            #pragma unroll
            for (int q = 0; q < 8; ++q) {
                int row = r * 16 + moff + q, col = ct * 16 + n;
                if (OutF) OutF[(size_t)row * ldof + col] = acc[r][q];
                if (OutG) OutG[(size_t)row * ldog + col] = acc[r][q];
            }
        }
    }
}

// ---------- weight prep: f32 [K,N] -> bf16 fragment layout, optional row scale g ----------
__global__ void prep_frag_kernel(const float* __restrict__ W, const float* __restrict__ g,
                                 __bf16* __restrict__ dst, int Kreal, int N, int ctiles) {
    int frag = blockIdx.x;                 // kt*ctiles + ct
    int ct = frag % ctiles, kt = frag / ctiles;
    int lane = threadIdx.x;                // 32 threads
    int col = ct * 16 + (lane & 15);
    int kb = kt * 32 + (lane >> 4) * 16;
    __bf16* p = dst + (((size_t)frag) * 32 + lane) * 16;
    #pragma unroll
    for (int j = 0; j < 16; ++j) {
        int k = kb + j;
        float w = 0.f;
        if (k < Kreal) {
            w = W[(size_t)k * N + col];
            if (g) w *= g[k];
        }
        p[j] = f2bf(w);
    }
}

// bias' = b1 + bvec @ W   (folds LN beta into the GEMM bias)
__global__ void bias_fold_kernel(const float* __restrict__ W, const float* __restrict__ bvec,
                                 const float* __restrict__ b1, float* __restrict__ dst,
                                 int K, int N) {
    int n = threadIdx.x + blockIdx.x * blockDim.x;
    if (n >= N) return;
    float s = b1[n];
    for (int k = 0; k < K; ++k) s += bvec[k] * W[(size_t)k * N + n];
    dst[n] = s;
}

struct KParams {
    const float *pt, *sh, *sem, *conf, *agree, *vis;
    const int* mask;
    const float *skB2, *svB2, *qB2, *uW2, *uB2;
    // prepared weights (bf16 fragment layout) + folded biases (f32), in d_ws
    const __bf16 *skW1p, *svW1p, *qW1p, *uW1p, *e2sWp, *skW2p, *svW2p, *qW2p;
    const float *skb1f, *svb1f, *qb1f, *ub1f, *e2sbf;
    float *o_le, *o_emb, *o_attn, *o_usage, *o_ent, *o_ma;
};

// LDS layout (dynamic, 294912 B total):
//  [0,98304)        semb   bf16 [128][384]   (LN-normalized sem, affine folded into weights)
//  [98304,163840)   h1     bf16 [128][256]
//  [163840,229376)  keyb   bf16 [128][256]
//  [229376,294912)  valbT  bf16 [256][128]   (value transposed)
// Phase-2 reuse of [0,163840):
#define OFF_QIN 0u        /* bf16 [32][544] : 34816 */
#define OFF_QH1 34816u    /* bf16 [32][256] : 16384 */
#define OFF_QRY 51200u    /* bf16 [32][256] : 16384 */
#define OFF_LOG 67584u    /* f32  [32][128] : 16384 */
#define OFF_ATT 83968u    /* bf16 [32][128] :  8192 */
#define OFF_EVF 92160u    /* f32  [32][256] : 32768 */
#define OFF_EVB 124928u   /* bf16 [32][256] : 16384 */
#define OFF_UIN 141312u   /* bf16 [32][288] : 18432 */
#define OFF_EMB 0u        /* f32  [32][384] : 49152 (after q-MLP done) */
#define OFF_UH1 0u        /* bf16 [32][256] : 16384 (after emb done)   */

__global__ void __launch_bounds__(512)
fused_encoder_kernel(KParams P) {
    extern __shared__ char smem[];
    __bf16* semb  = (__bf16*)smem;
    __bf16* h1    = (__bf16*)(smem + 98304);
    __bf16* keyb  = (__bf16*)(smem + 163840);
    __bf16* valbT = (__bf16*)(smem + 229376);
    __bf16* qin    = (__bf16*)(smem + OFF_QIN);
    __bf16* qh1    = (__bf16*)(smem + OFF_QH1);
    __bf16* queryb = (__bf16*)(smem + OFF_QRY);
    float*  logit  = (float*)(smem + OFF_LOG);
    __bf16* attnb  = (__bf16*)(smem + OFF_ATT);
    float*  evidf  = (float*)(smem + OFF_EVF);
    __bf16* evidb  = (__bf16*)(smem + OFF_EVB);
    __bf16* uin    = (__bf16*)(smem + OFF_UIN);
    float*  embf   = (float*)(smem + OFF_EMB);
    __bf16* uh1    = (__bf16*)(smem + OFF_UH1);

    __shared__ float s_cm, s_am, s_vm;
    __shared__ float s_ent[Hc], s_ma[Hc];

    const int bm   = blockIdx.x;     // b*M + m
    const int b    = bm >> 9;        // / 512
    const int lane = threadIdx.x & 31;
    const int wave = threadIdx.x >> 5;
    const int nwaves = 16;

    // ---- Phase A: LN-normalize sem -> semb; reduce conf/agree/vis means ----
    for (int r = wave; r < Tc; r += nwaves) {
        float xv[12], s = 0.f, sq = 0.f;
        #pragma unroll
        for (int i = 0; i < 12; ++i) {
            float x = P.sem[((size_t)bm * Tc + r) * SEMc + lane + i * 32];
            x = __builtin_isfinite(x) ? x : 0.f;
            xv[i] = x; s += x; sq += x * x;
        }
        s = wred_sum(s); sq = wred_sum(sq);
        float mu = s * (1.f / SEMc);
        float var = sq * (1.f / SEMc) - mu * mu;
        float rstd = rsqrtf(var + 1e-5f);
        #pragma unroll
        for (int i = 0; i < 12; ++i)
            semb[(size_t)r * SEMc + lane + i * 32] = f2bf((xv[i] - mu) * rstd);
    }
    if (wave == 0) {
        float sc = 0.f, sa = 0.f, sv = 0.f;
        #pragma unroll
        for (int i = 0; i < 4; ++i) {
            int t = lane + i * 32;
            sc += clamp01(P.conf[(size_t)bm * Tc + t]);
            sa += clamp01(P.agree[(size_t)bm * Tc + t]);
            sv += clamp01(P.vis[(size_t)bm * Tc + t]);
        }
        sc = wred_sum(sc); sa = wred_sum(sa); sv = wred_sum(sv);
        if (lane == 0) { s_cm = sc / Tc; s_am = sa / Tc; s_vm = sv / Tc; }
    }
    __syncthreads();

    // ---- key / value MLPs ----
    gemm_prep<8>(semb, SEMc, P.skW1p, 12, P.skb1f, 16, 1,
                 h1, HIDc, false, nullptr, 0, nullptr, 0, wave, nwaves, lane);
    __syncthreads();
    gemm_prep<8>(h1, HIDc, P.skW2p, 8, P.skB2, 16, 0,
                 keyb, HIDc, false, nullptr, 0, nullptr, 0, wave, nwaves, lane);
    __syncthreads();
    gemm_prep<8>(semb, SEMc, P.svW1p, 12, P.svb1f, 16, 1,
                 h1, HIDc, false, nullptr, 0, nullptr, 0, wave, nwaves, lane);
    __syncthreads();
    gemm_prep<8>(h1, HIDc, P.svW2p, 8, P.svB2, 16, 0,
                 valbT, Tc, true, nullptr, 0, nullptr, 0, wave, nwaves, lane);
    __syncthreads();

    // ---- q_in [32,515] -> normalized -> qin bf16 [32,544] (affine folded) ----
    for (int h = wave * 2; h < wave * 2 + 2; ++h) {
        float vv[17], s = 0.f, sq = 0.f; int c = 0;
        for (int k = lane; k < 515; k += 32) {
            float x;
            if (k < 256)      x = P.pt[(size_t)bm * HIDc + k];
            else if (k < 512) x = P.sh[((size_t)b * Hc + h) * HIDc + (k - 256)];
            else if (k == 512) x = (float)h * (1.f / (Hc - 1));
            else if (k == 513) x = s_cm;
            else               x = s_am * s_vm;
            vv[c++] = x; s += x; sq += x * x;
        }
        s = wred_sum(s); sq = wred_sum(sq);
        float mu = s * (1.f / 515.f);
        float var = sq * (1.f / 515.f) - mu * mu;
        float rstd = rsqrtf(var + 1e-5f);
        c = 0;
        for (int k = lane; k < 515; k += 32)
            qin[h * 544 + k] = f2bf((vv[c++] - mu) * rstd);
        for (int k = lane; k < 544; k += 32)
            if (k >= 515) qin[h * 544 + k] = f2bf(0.f);
    }
    __syncthreads();

    // ---- query MLP ----
    gemm_prep<2>(qin, 544, P.qW1p, 17, P.qb1f, 16, 1,
                 qh1, HIDc, false, nullptr, 0, nullptr, 0, wave, nwaves, lane);
    __syncthreads();
    gemm_prep<2>(qh1, HIDc, P.qW2p, 8, P.qB2, 16, 0,
                 queryb, HIDc, false, nullptr, 0, nullptr, 0, wave, nwaves, lane);
    __syncthreads();

    // ---- logits = q @ key^T / 16 + biases, mask ----
    if (wave < 8) {
        const f32x8 zero = {0.f, 0.f, 0.f, 0.f, 0.f, 0.f, 0.f, 0.f};
        f32x8 acc[2] = {zero, zero};
        for (int k0 = 0; k0 < HIDc; k0 += 32) {
            bf16x16 bf = load_bT_lds(keyb, HIDc, k0, wave * 16, lane);
            #pragma unroll
            for (int r = 0; r < 2; ++r) {
                bf16x16 af = load_a(queryb, HIDc, r * 16, k0, lane);
                acc[r] = __builtin_amdgcn_wmma_f32_16x16x32_bf16(
                    false, af, false, bf, (short)0, acc[r], false, false);
            }
        }
        int n = lane & 15, moff = (lane >> 4) * 8;
        #pragma unroll
        for (int r = 0; r < 2; ++r) {
            #pragma unroll
            for (int q = 0; q < 8; ++q) {
                int row = r * 16 + moff + q;
                int t = wave * 16 + n;
                float cf = clamp01(P.conf[(size_t)bm * Tc + t]);
                float ag = clamp01(P.agree[(size_t)bm * Tc + t]);
                float x = acc[r][q] * 0.0625f + 1.5f * cf + 1.5f * ag
                        + 0.25f * ((float)t * (1.f / (Tc - 1)));
                if (!P.mask[(size_t)bm * Tc + t]) x = -10000.f;
                logit[row * Tc + t] = x;
            }
        }
    }
    __syncthreads();

    // ---- softmax + mask-renorm + entropy/max ----
    for (int h = wave * 2; h < wave * 2 + 2; ++h) {
        float lv[4], mx = -1e30f;
        #pragma unroll
        for (int i = 0; i < 4; ++i) { lv[i] = logit[h * Tc + lane + i * 32]; mx = fmaxf(mx, lv[i]); }
        mx = wred_max(mx);
        float s = 0.f;
        #pragma unroll
        for (int i = 0; i < 4; ++i) { lv[i] = expf(lv[i] - mx); s += lv[i]; }
        s = wred_sum(s);
        float inv = 1.f / s;
        float a[4], s2 = 0.f;
        #pragma unroll
        for (int i = 0; i < 4; ++i) {
            int t = lane + i * 32;
            float mk = P.mask[(size_t)bm * Tc + t] ? 1.f : 0.f;
            a[i] = lv[i] * inv * mk; s2 += a[i];
        }
        s2 = wred_sum(s2);
        float r2 = 1.f / fmaxf(s2, 1e-6f);
        float ent = 0.f, ma = 0.f;
        #pragma unroll
        for (int i = 0; i < 4; ++i) {
            a[i] *= r2;
            int t = lane + i * 32;
            P.o_attn[((size_t)bm * Hc + h) * Tc + t] = a[i];
            attnb[h * Tc + t] = f2bf(a[i]);
            float pa = fmaxf(a[i], 1e-8f);
            ent -= pa * logf(pa);
            ma = fmaxf(ma, a[i]);
        }
        ent = wred_sum(ent); ma = wred_max(ma);
        if (lane == 0) {
            float e = ent * (1.f / logf((float)Tc));
            P.o_ent[(size_t)bm * Hc + h] = e; s_ent[h] = e;
            P.o_ma[(size_t)bm * Hc + h] = ma; s_ma[h] = ma;
        }
    }
    __syncthreads();

    // ---- evidence = attn @ value (via value^T) -> evidf (LDS) + o_le (global) ----
    gemm_ldsBT<2>(attnb, Tc, valbT, Tc, Tc, 16,
                  evidf, HIDc, P.o_le + (size_t)bm * Hc * HIDc, HIDc, wave, nwaves, lane);
    __syncthreads();

    // ---- LN(evidence) -> evidb (normalized; e2s affine folded into weights) ----
    for (int h = wave * 2; h < wave * 2 + 2; ++h) {
        float xv[8], s = 0.f, sq = 0.f;
        #pragma unroll
        for (int i = 0; i < 8; ++i) { float x = evidf[h * HIDc + lane + i * 32]; xv[i] = x; s += x; sq += x * x; }
        s = wred_sum(s); sq = wred_sum(sq);
        float mu = s * (1.f / HIDc), var = sq * (1.f / HIDc) - mu * mu;
        float rstd = rsqrtf(var + 1e-5f);
        #pragma unroll
        for (int i = 0; i < 8; ++i) {
            int k = lane + i * 32;
            evidb[h * HIDc + k] = f2bf((xv[i] - mu) * rstd);
        }
    }
    __syncthreads();

    // ---- emb = LN(evid) @ e2sW' + bias' -> embf; L2-normalize -> o_emb ----
    gemm_prep<2>(evidb, HIDc, P.e2sWp, 8, P.e2sbf, 24, 0,
                 nullptr, 0, false, embf, SEMc, nullptr, 0, wave, nwaves, lane);
    __syncthreads();
    for (int h = wave * 2; h < wave * 2 + 2; ++h) {
        float s = 0.f;
        #pragma unroll
        for (int i = 0; i < 12; ++i) { float x = embf[h * SEMc + lane + i * 32]; s += x * x; }
        s = wred_sum(s);
        float rn = 1.f / fmaxf(sqrtf(s), 1e-12f);
        #pragma unroll
        for (int i = 0; i < 12; ++i) {
            int k = lane + i * 32;
            P.o_emb[((size_t)bm * Hc + h) * SEMc + k] = embf[h * SEMc + k] * rn;
        }
    }
    __syncthreads();

    // ---- usage_in [32,260] -> normalized -> uin bf16 [32,288] ----
    for (int h = wave * 2; h < wave * 2 + 2; ++h) {
        float vv[9], s = 0.f, sq = 0.f; int c = 0;
        for (int k = lane; k < 260; k += 32) {
            float x;
            if (k < 256)       x = evidf[h * HIDc + k];
            else if (k == 256) x = s_ent[h];
            else if (k == 257) x = s_ma[h];
            else if (k == 258) x = s_cm;
            else               x = s_am;
            vv[c++] = x; s += x; sq += x * x;
        }
        s = wred_sum(s); sq = wred_sum(sq);
        float mu = s * (1.f / 260.f), var = sq * (1.f / 260.f) - mu * mu;
        float rstd = rsqrtf(var + 1e-5f);
        c = 0;
        for (int k = lane; k < 260; k += 32)
            uin[h * 288 + k] = f2bf((vv[c++] - mu) * rstd);
        for (int k = lane; k < 288; k += 32)
            if (k >= 260) uin[h * 288 + k] = f2bf(0.f);
    }
    __syncthreads();

    // ---- usage hidden layer (WMMA) ----
    gemm_prep<2>(uin, 288, P.uW1p, 9, P.ub1f, 16, 1,
                 uh1, HIDc, false, nullptr, 0, nullptr, 0, wave, nwaves, lane);
    __syncthreads();

    // ---- usage output: sigmoid(uh1 @ uW2 + b2) ----
    for (int h = wave * 2; h < wave * 2 + 2; ++h) {
        float s = 0.f;
        #pragma unroll
        for (int i = 0; i < 8; ++i) {
            int k = lane + i * 32;
            s += bf2f(uh1[h * HIDc + k]) * P.uW2[k];
        }
        s = wred_sum(s);
        if (lane == 0) {
            float z = s + P.uB2[0];
            P.o_usage[(size_t)bm * Hc + h] = 1.f / (1.f + expf(-z));
        }
    }
}

// ---- Kernel 2: unit_evidence / unit_usage reductions over m ----
__global__ void __launch_bounds__(256)
unit_reduce_kernel(const float* assign, const float* usage, const float* le,
                   float* o_ue, float* o_uu) {
    extern __shared__ char sm2[];
    float* s_as  = (float*)sm2;          // [512][64]
    float* s_us  = s_as + Mc * Uc;       // [512]
    float* s_den = s_us + Mc;            // [64]
    const int b = blockIdx.x >> 5;       // / 32
    const int h = blockIdx.x & 31;
    const int tid = threadIdx.x;

    for (int i = tid; i < Mc * Uc; i += 256) s_as[i] = assign[(size_t)b * Mc * Uc + i];
    for (int m = tid; m < Mc; m += 256)      s_us[m] = usage[((size_t)b * Mc + m) * Hc + h];
    __syncthreads();

    if (tid < Uc) {
        float d = 0.f;
        for (int m = 0; m < Mc; ++m) d += s_as[m * Uc + tid];
        s_den[tid] = fmaxf(d, 1e-6f);
    }

    float acc[Uc];
    #pragma unroll
    for (int u = 0; u < Uc; ++u) acc[u] = 0.f;
    const int d = tid;
    for (int m = 0; m < Mc; ++m) {
        float x = le[(((size_t)b * Mc + m) * Hc + h) * HIDc + d];
        const float* ar = s_as + m * Uc;
        #pragma unroll
        for (int u = 0; u < Uc; ++u) acc[u] += ar[u] * x;
    }
    __syncthreads();
    for (int u = 0; u < Uc; ++u)
        o_ue[(((size_t)b * Uc + u) * Hc + h) * HIDc + d] = acc[u] / s_den[u];
    if (tid < Uc) {
        float a2 = 0.f;
        for (int m = 0; m < Mc; ++m) a2 += s_as[m * Uc + tid] * s_us[m];
        float v = a2 / s_den[tid];
        o_uu[((size_t)b * Uc + tid) * Hc + h] = fminf(fmaxf(v, 0.f), 1.f);
    }
}

extern "C" void kernel_launch(void* const* d_in, const int* in_sizes, int n_in,
                              void* d_out, int out_size, void* d_ws, size_t ws_size,
                              hipStream_t stream) {
    (void)in_sizes; (void)n_in; (void)ws_size; (void)out_size;
    const float* pt    = (const float*)d_in[0];
    const float* sh    = (const float*)d_in[1];
    const float* sem   = (const float*)d_in[2];
    const int*   mask  = (const int*)  d_in[3];
    const float* conf  = (const float*)d_in[4];
    const float* agree = (const float*)d_in[5];
    const float* vis   = (const float*)d_in[6];
    const float* assign= (const float*)d_in[7];
    // params tree-flattened with sorted dict keys:
    // e2s{W,b,bb,g}, q{W1,W2,b,b1,b2,g}, sk{...}, sv{...}, u{...}
    const float* e2sW = (const float*)d_in[8];  const float* e2sB = (const float*)d_in[9];
    const float* e2sBB= (const float*)d_in[10]; const float* e2sG = (const float*)d_in[11];
    const float* qW1  = (const float*)d_in[12]; const float* qW2  = (const float*)d_in[13];
    const float* qB   = (const float*)d_in[14]; const float* qB1  = (const float*)d_in[15];
    const float* qB2  = (const float*)d_in[16]; const float* qG   = (const float*)d_in[17];
    const float* skW1 = (const float*)d_in[18]; const float* skW2 = (const float*)d_in[19];
    const float* skB  = (const float*)d_in[20]; const float* skB1 = (const float*)d_in[21];
    const float* skB2 = (const float*)d_in[22]; const float* skG  = (const float*)d_in[23];
    const float* svW1 = (const float*)d_in[24]; const float* svW2 = (const float*)d_in[25];
    const float* svB  = (const float*)d_in[26]; const float* svB1 = (const float*)d_in[27];
    const float* svB2 = (const float*)d_in[28]; const float* svG  = (const float*)d_in[29];
    const float* uW1  = (const float*)d_in[30]; const float* uW2  = (const float*)d_in[31];
    const float* uB   = (const float*)d_in[32]; const float* uB1  = (const float*)d_in[33];
    const float* uB2  = (const float*)d_in[34]; const float* uG   = (const float*)d_in[35];

    // ---- workspace layout (bytes) ----
    char* w = (char*)d_ws;
    __bf16* skW1p = (__bf16*)(w + 0);        // 12*16*512*2 = 196608
    __bf16* svW1p = (__bf16*)(w + 196608);   // 196608
    __bf16* qW1p  = (__bf16*)(w + 393216);   // 17*16*512*2 = 278528
    __bf16* uW1p  = (__bf16*)(w + 671744);   //  9*16*512*2 = 147456
    __bf16* e2sWp = (__bf16*)(w + 819200);   //  8*24*512*2 = 196608
    __bf16* skW2p = (__bf16*)(w + 1015808);  //  8*16*512*2 = 131072
    __bf16* svW2p = (__bf16*)(w + 1146880);  // 131072
    __bf16* qW2p  = (__bf16*)(w + 1277952);  // 131072
    float*  skb1f = (float*)(w + 1409024);   // 1024
    float*  svb1f = (float*)(w + 1410048);
    float*  qb1f  = (float*)(w + 1411072);
    float*  ub1f  = (float*)(w + 1412096);
    float*  e2sbf = (float*)(w + 1413120);   // 1536

    // ---- prep: scaled bf16 fragment weights + folded biases ----
    prep_frag_kernel<<<dim3(12 * 16), dim3(32), 0, stream>>>(skW1, skG, skW1p, 384, 256, 16);
    prep_frag_kernel<<<dim3(12 * 16), dim3(32), 0, stream>>>(svW1, svG, svW1p, 384, 256, 16);
    prep_frag_kernel<<<dim3(17 * 16), dim3(32), 0, stream>>>(qW1,  qG,  qW1p,  515, 256, 16);
    prep_frag_kernel<<<dim3( 9 * 16), dim3(32), 0, stream>>>(uW1,  uG,  uW1p,  260, 256, 16);
    prep_frag_kernel<<<dim3( 8 * 24), dim3(32), 0, stream>>>(e2sW, e2sG, e2sWp, 256, 384, 24);
    prep_frag_kernel<<<dim3( 8 * 16), dim3(32), 0, stream>>>(skW2, nullptr, skW2p, 256, 256, 16);
    prep_frag_kernel<<<dim3( 8 * 16), dim3(32), 0, stream>>>(svW2, nullptr, svW2p, 256, 256, 16);
    prep_frag_kernel<<<dim3( 8 * 16), dim3(32), 0, stream>>>(qW2,  nullptr, qW2p,  256, 256, 16);
    bias_fold_kernel<<<dim3(1), dim3(256), 0, stream>>>(skW1, skB, skB1, skb1f, 384, 256);
    bias_fold_kernel<<<dim3(1), dim3(256), 0, stream>>>(svW1, svB, svB1, svb1f, 384, 256);
    bias_fold_kernel<<<dim3(1), dim3(256), 0, stream>>>(qW1,  qB,  qB1,  qb1f,  515, 256);
    bias_fold_kernel<<<dim3(1), dim3(256), 0, stream>>>(uW1,  uB,  uB1,  ub1f,  260, 256);
    bias_fold_kernel<<<dim3(2), dim3(192), 0, stream>>>(e2sW, e2sB, e2sBB, e2sbf, 256, 384);

    // ---- outputs ----
    float* out = (float*)d_out;
    const size_t n_le   = (size_t)Bc * Mc * Hc * HIDc;
    const size_t n_emb  = (size_t)Bc * Mc * Hc * SEMc;
    const size_t n_attn = (size_t)Bc * Mc * Hc * Tc;
    const size_t n_us   = (size_t)Bc * Mc * Hc;
    const size_t n_ue   = (size_t)Bc * Uc * Hc * HIDc;
    const size_t n_uu   = (size_t)Bc * Uc * Hc;

    KParams P;
    P.pt = pt; P.sh = sh; P.sem = sem; P.conf = conf; P.agree = agree; P.vis = vis;
    P.mask = mask;
    P.skB2 = skB2; P.svB2 = svB2; P.qB2 = qB2; P.uW2 = uW2; P.uB2 = uB2;
    P.skW1p = skW1p; P.svW1p = svW1p; P.qW1p = qW1p; P.uW1p = uW1p; P.e2sWp = e2sWp;
    P.skW2p = skW2p; P.svW2p = svW2p; P.qW2p = qW2p;
    P.skb1f = skb1f; P.svb1f = svb1f; P.qb1f = qb1f; P.ub1f = ub1f; P.e2sbf = e2sbf;
    P.o_le    = out;
    P.o_emb   = P.o_le + n_le;
    P.o_attn  = P.o_emb + n_emb;
    P.o_usage = P.o_attn + n_attn;
    float* o_ue = P.o_usage + n_us;
    float* o_uu = o_ue + n_ue;
    P.o_ent   = o_uu + n_uu;
    P.o_ma    = P.o_ent + n_us;

    fused_encoder_kernel<<<dim3(Bc * Mc), dim3(512), 294912, stream>>>(P);
    unit_reduce_kernel<<<dim3(Bc * Hc), dim3(256),
                         (Mc * Uc + Mc + Uc) * sizeof(float), stream>>>(
        assign, P.o_usage, P.o_le, o_ue, o_uu);
}
